// GraphMultiAttn_42984032698881
// MI455X (gfx1250) — matrix operand order repaired and checked by reading, not compiled, not run on hardware
//
#include <hip/hip_runtime.h>
#include <hip/hip_bf16.h>
#include <math.h>
#include <stdint.h>

// ---------------- problem constants (from reference) ----------------
#define NN   10000
#define LL   1024
#define DD   256
#define HH   4
#define NCLSC 2
#define EE   80000
#define HCO  256

// ---------------- WMMA types ----------------
typedef __bf16 bf16_t;
typedef __attribute__((ext_vector_type(16))) __bf16 v16bf;
typedef __attribute__((ext_vector_type(8)))  float  v8f;

union V8BF { uint4 u; bf16_t b[8]; unsigned short s[8]; };

// ---------------- helpers ----------------
__device__ __forceinline__ unsigned short f32_to_bf16_bits(float f) {
  unsigned int u = __float_as_uint(f);
  unsigned int lsb = (u >> 16) & 1u;
  u += 0x7fffu + lsb;                      // round-to-nearest-even
  return (unsigned short)(u >> 16);
}

__device__ __forceinline__ float gelu_exact(float v) {
  return 0.5f * v * (1.0f + erff(v * 0.70710678118654752f));
}

__device__ __forceinline__ void atomicMaxFloat(float* addr, float v) {
  // valid when buffer initialized to -inf
  if (v >= 0.0f) atomicMax((int*)addr, __float_as_int(v));
  else           atomicMin((unsigned int*)addr, __float_as_uint(v));
}

// ---------------- elementwise kernels ----------------
__global__ void cast_f32_to_bf16_kernel(const float* __restrict__ src,
                                        unsigned short* __restrict__ dst, int n) {
  int i = blockIdx.x * blockDim.x + threadIdx.x;
  if (i < n) dst[i] = f32_to_bf16_bits(src[i]);
}

__global__ void copy_f32_kernel(const float* __restrict__ s, float* __restrict__ d, int n) {
  int i = blockIdx.x * blockDim.x + threadIdx.x;
  if (i < n) d[i] = s[i];
}

__global__ void fill_f32_kernel(float* p, float v, int n) {
  int i = blockIdx.x * blockDim.x + threadIdx.x;
  if (i < n) p[i] = v;
}

__global__ void init_seg_kernel(float* m, float* s, int n) {
  int i = blockIdx.x * blockDim.x + threadIdx.x;
  if (i < n) { m[i] = -INFINITY; s[i] = 0.0f; }
}

__global__ void gelu2_kernel(float* x, int n) {
  int i = blockIdx.x * blockDim.x + threadIdx.x;
  if (i < n) x[i] = gelu_exact(gelu_exact(x[i]));
}

__global__ void gate_kernel(float* gv, const float* __restrict__ gu, int n) {
  int i = blockIdx.x * blockDim.x + threadIdx.x;
  if (i < n) gv[i] = tanhf(gv[i]) * (1.0f / (1.0f + expf(-gu[i])));
}

// ---------------- bf16 WMMA GEMM: C[M,N] = A[M,K] @ B[K,N] + bias ----------------
#define BM 128
#define BN 64
#define BK 32
#define LDSROW (BK + 8)

__global__ __launch_bounds__(256, 2)
void gemm_bf16_wmma_kernel(const unsigned short* __restrict__ A,   // M x K (bf16 bits)
                           const unsigned short* __restrict__ B,   // K x N (bf16 bits)
                           const float* __restrict__ bias,         // N or nullptr
                           float* __restrict__ C,                  // M x N
                           int M, int N, int K)
{
  // single shared block; LDS-relative addresses taken from low 32 bits of flat addr
  __shared__ unsigned short smem[BM * LDSROW + BN * LDSROW];
  unsigned short (*sA)[LDSROW]  = (unsigned short (*)[LDSROW])smem;                 // A tile, row-major
  unsigned short (*sBt)[LDSROW] = (unsigned short (*)[LDSROW])(smem + BM * LDSROW); // B tile, N-major

  const int tid  = threadIdx.x;
  const int wave = tid >> 5;        // 0..7
  const int lane = tid & 31;
  const int wm   = wave >> 1;       // 0..3 -> 32-row slab
  const int wn   = wave & 1;        // 0..1 -> 32-col slab
  const int half = lane >> 4;       // 0/1
  const int l16  = lane & 15;

  const int blockRow = blockIdx.y * BM;
  const int blockCol = blockIdx.x * BN;

  v8f acc[2][2];
  #pragma unroll
  for (int a = 0; a < 2; ++a)
    #pragma unroll
    for (int b = 0; b < 2; ++b)
      acc[a][b] = (v8f)(0.0f);

  for (int k0 = 0; k0 < K; k0 += BK) {
    // --- stage A tile via async global->LDS DMA (ASYNCcnt path) ---
    // BM*BK/8 = 512 lane-transfers of 16B; 2 wave-wide async instrs per wave.
    // Lanes with gr >= M stay masked; stale LDS rows only feed C rows >= M
    // which are guarded at the store, so no zero-fill is required.
    #pragma unroll
    for (int v = tid; v < (BM * BK) / 8; v += 256) {
      int r  = v >> 2;              // BK/8 = 4 vec8 per row
      int c8 = (v & 3) * 8;
      int gr = blockRow + r;
      if (gr < M) {
        unsigned ldsoff = (unsigned)(uintptr_t)&sA[r][c8];
        unsigned long long gaddr =
            (unsigned long long)(uintptr_t)(A + (size_t)gr * K + k0 + c8);
        asm volatile("global_load_async_to_lds_b128 %0, %1, off"
                     :: "v"(ldsoff), "v"(gaddr) : "memory");
      }
    }
    // --- stage B tile transposed: BK*BN/8 = 256 vec8 reads, scatter to N-major ---
    {
      int r  = tid >> 3;            // k row 0..31 (BN/8 = 8 vec8 per row)
      int c8 = (tid & 7) * 8;
      V8BF ld;
      ld.u = *(const uint4*)(B + (size_t)(k0 + r) * N + blockCol + c8);
      #pragma unroll
      for (int j = 0; j < 8; ++j) sBt[c8 + j][r] = ld.s[j];
    }
    // prefetch next B k-rows into cache
    if (k0 + BK < K) {
      __builtin_prefetch(B + (size_t)(k0 + BK + (tid >> 3)) * N + blockCol, 0, 0);
    }
    // wait for this wave's async DMA, then make LDS visible workgroup-wide
    asm volatile("s_wait_asynccnt 0x0" ::: "memory");
    __syncthreads();

    // --- build fragments (contiguous 16B LDS loads per ISA lane layout) ---
    v16bf af[2], bf[2];
    #pragma unroll
    for (int mt = 0; mt < 2; ++mt) {
      int arow = wm * 32 + mt * 16 + l16;
      V8BF lo, hi;
      lo.u = *(const uint4*)&sA[arow][half * 8];        // K = half*8 .. +7
      hi.u = *(const uint4*)&sA[arow][16 + half * 8];   // K = 16+half*8 .. +7
      #pragma unroll
      for (int j = 0; j < 8; ++j) { af[mt][j] = lo.b[j]; af[mt][8 + j] = hi.b[j]; }
    }
    #pragma unroll
    for (int nt = 0; nt < 2; ++nt) {
      int bcol = wn * 32 + nt * 16 + l16;
      V8BF lo, hi;
      lo.u = *(const uint4*)&sBt[bcol][half * 8];       // K = half*8 .. +7
      hi.u = *(const uint4*)&sBt[bcol][16 + half * 8];  // K = 16+half*8 .. +7
      #pragma unroll
      for (int j = 0; j < 8; ++j) { bf[nt][j] = lo.b[j]; bf[nt][8 + j] = hi.b[j]; }
    }

    // --- 4 WMMAs per wave per k-step ---
    #pragma unroll
    for (int mt = 0; mt < 2; ++mt)
      #pragma unroll
      for (int nt = 0; nt < 2; ++nt)
        acc[mt][nt] = __builtin_amdgcn_wmma_f32_16x16x32_bf16(
            false, af[mt], false, bf[nt], (short)0, acc[mt][nt], false, false);

    __syncthreads();
  }

  // --- store: D layout N = lane&15, M = i + 8*(lane>>4) ---
  #pragma unroll
  for (int mt = 0; mt < 2; ++mt) {
    #pragma unroll
    for (int nt = 0; nt < 2; ++nt) {
      int col = blockCol + wn * 32 + nt * 16 + l16;
      float bv = (bias != nullptr) ? bias[col] : 0.0f;
      #pragma unroll
      for (int i = 0; i < 8; ++i) {
        int row = blockRow + wm * 32 + mt * 16 + half * 8 + i;
        if (row < M) C[(size_t)row * N + col] = acc[mt][nt][i] + bv;
      }
    }
  }
}

// ---------------- edge attention kernels ----------------
// one wave per (edge, head): logits[e,h] = scale * dot(q[dst], k[src])
__global__ void edge_logits_kernel(const float* __restrict__ q, const float* __restrict__ k,
                                   const int* __restrict__ srcI, const int* __restrict__ dstI,
                                   float* __restrict__ logits, int E, int heads, int c, float scale)
{
  int task = blockIdx.x * 8 + (threadIdx.x >> 5);
  int lane = threadIdx.x & 31;
  if (task >= E * heads) return;
  int e = task / heads, h = task - e * heads;
  int hc = heads * c;
  const float* qp = q + (size_t)dstI[e] * hc + h * c;
  const float* kp = k + (size_t)srcI[e] * hc + h * c;
  float acc = 0.0f;
  for (int j = lane; j < c; j += 32) acc += qp[j] * kp[j];
  #pragma unroll
  for (int off = 16; off > 0; off >>= 1) acc += __shfl_xor(acc, off, 32);
  if (lane == 0) logits[(size_t)e * heads + h] = acc * scale;
}

__global__ void seg_max_kernel(const float* __restrict__ logits, const int* __restrict__ dstI,
                               float* __restrict__ m, int E, int heads)
{
  int i = blockIdx.x * blockDim.x + threadIdx.x;
  if (i >= E * heads) return;
  int e = i / heads, h = i - e * heads;
  atomicMaxFloat(&m[(size_t)dstI[e] * heads + h], logits[i]);
}

__global__ void m_fix_kernel(float* m, int n) {
  int i = blockIdx.x * blockDim.x + threadIdx.x;
  if (i < n) { float v = m[i]; m[i] = isfinite(v) ? v : 0.0f; }
}

__global__ void edge_exp_kernel(const float* __restrict__ logits, const float* __restrict__ m,
                                const int* __restrict__ dstI, float* __restrict__ ew,
                                float* __restrict__ ssum, int E, int heads)
{
  int i = blockIdx.x * blockDim.x + threadIdx.x;
  if (i >= E * heads) return;
  int e = i / heads, h = i - e * heads;
  float v = expf(logits[i] - m[(size_t)dstI[e] * heads + h]);
  ew[i] = v;
  atomicAdd(&ssum[(size_t)dstI[e] * heads + h], v);
}

// out[dst] += alpha * v[src]  (out already holds x@Ws + bs)
__global__ void edge_aggregate_kernel(const float* __restrict__ ew, const float* __restrict__ ssum,
                                      const float* __restrict__ vmat,
                                      const int* __restrict__ srcI, const int* __restrict__ dstI,
                                      float* __restrict__ out, int heads, int c)
{
  int e = blockIdx.x;
  int s = srcI[e], d = dstI[e];
  int hc = heads * c;
  for (int j = threadIdx.x; j < hc; j += blockDim.x) {
    int h = j / c;
    float alpha = ew[(size_t)e * heads + h] / (ssum[(size_t)d * heads + h] + 1e-16f);
    atomicAdd(&out[(size_t)d * hc + j], alpha * vmat[(size_t)s * hc + j]);
  }
}

// ---------------- graph layernorm (global mean/var) ----------------
__global__ void reduce_sum_kernel(const float* __restrict__ x, float* scal, int n) {
  __shared__ float sm[256];
  float a = 0.0f;
  for (int i = blockIdx.x * blockDim.x + threadIdx.x; i < n; i += gridDim.x * blockDim.x) a += x[i];
  sm[threadIdx.x] = a; __syncthreads();
  for (int s = 128; s > 0; s >>= 1) {
    if (threadIdx.x < s) sm[threadIdx.x] += sm[threadIdx.x + s];
    __syncthreads();
  }
  if (threadIdx.x == 0) atomicAdd(&scal[0], sm[0]);
}

__global__ void reduce_sq_kernel(const float* __restrict__ x, float* scal, int n) {
  __shared__ float sm[256];
  float mu = scal[0] / (float)n;
  float a = 0.0f;
  for (int i = blockIdx.x * blockDim.x + threadIdx.x; i < n; i += gridDim.x * blockDim.x) {
    float d = x[i] - mu; a += d * d;
  }
  sm[threadIdx.x] = a; __syncthreads();
  for (int s = 128; s > 0; s >>= 1) {
    if (threadIdx.x < s) sm[threadIdx.x] += sm[threadIdx.x + s];
    __syncthreads();
  }
  if (threadIdx.x == 0) atomicAdd(&scal[1], sm[0]);
}

// hcur = gelu( (hnew - mu)/sqrt(var+eps) * w[col] + b[col] ) + hcur
__global__ void ln_apply_kernel(const float* __restrict__ hnew, float* __restrict__ hcur,
                                const float* __restrict__ w, const float* __restrict__ b,
                                const float* __restrict__ scal, int n, int Lc)
{
  int i = blockIdx.x * blockDim.x + threadIdx.x;
  if (i >= n) return;
  float mu  = scal[0] / (float)n;
  float var = scal[1] / (float)n;
  float inv = rsqrtf(var + 1e-5f);
  int col = i % Lc;
  float o = (hnew[i] - mu) * inv * w[col] + b[col];
  hcur[i] = gelu_exact(o) + hcur[i];
}

// ---------------- pooling kernels ----------------
__global__ void glog_kernel(const float* __restrict__ g, const float* __restrict__ wp,
                            float* __restrict__ glog, int n, int d)
{
  int i = blockIdx.x * blockDim.x + threadIdx.x;
  if (i >= n) return;
  const float* gp = g + (size_t)i * d;
  float a0 = 0.0f, a1 = 0.0f;
  for (int j = 0; j < d; ++j) {
    float gv = gp[j];
    a0 += gv * wp[j * NCLSC + 0];
    a1 += gv * wp[j * NCLSC + 1];
  }
  glog[(size_t)i * NCLSC + 0] = a0;
  glog[(size_t)i * NCLSC + 1] = a1;
}

__global__ void softmax_init_kernel(float* scal) {
  scal[2] = -INFINITY; scal[3] = -INFINITY; scal[4] = 0.0f; scal[5] = 0.0f;
}

__global__ void colmax_kernel(const float* __restrict__ glog, float* scal, int n) {
  int i = blockIdx.x * blockDim.x + threadIdx.x;
  if (i >= n) return;
  atomicMaxFloat(&scal[2 + (i & 1)], glog[i]);
}

__global__ void colexp_kernel(const float* __restrict__ glog, float* __restrict__ attn,
                              float* scal, int n) {
  int i = blockIdx.x * blockDim.x + threadIdx.x;
  if (i >= n) return;
  float v = expf(glog[i] - scal[2 + (i & 1)]);
  attn[i] = v;
  atomicAdd(&scal[4 + (i & 1)], v);
}

__global__ void coldiv_kernel(float* attn, const float* scal, int n) {
  int i = blockIdx.x * blockDim.x + threadIdx.x;
  if (i < n) attn[i] /= scal[4 + (i & 1)];
}

// y[cls, j] += sum over node chunk of attn[n,cls]*h[n,j]; grid = (chunks, NCLS), block = 256 (=D)
__global__ void yout_kernel(const float* __restrict__ attn, const float* __restrict__ h,
                            float* __restrict__ y, int chunk)
{
  int cls = blockIdx.y;
  int j = threadIdx.x;
  int n0 = blockIdx.x * chunk;
  float acc = 0.0f;
  for (int n = n0; n < n0 + chunk; ++n)
    acc += attn[(size_t)n * NCLSC + cls] * h[(size_t)n * DD + j];
  atomicAdd(&y[cls * DD + j], acc);
}

__global__ void aout_kernel(const float* __restrict__ attn, const int* __restrict__ label,
                            float* __restrict__ out, int n) {
  int i = blockIdx.x * blockDim.x + threadIdx.x;
  if (i < n) out[i] = attn[(size_t)i * NCLSC + label[0]];
}

// ---------------- host orchestration ----------------
static inline void launch_gemm(const unsigned short* A, const unsigned short* B,
                               const float* bias, float* C, int M, int N, int K,
                               hipStream_t stream) {
  dim3 grid(N / BN, (M + BM - 1) / BM);
  gemm_bf16_wmma_kernel<<<grid, 256, 0, stream>>>(A, B, bias, C, M, N, K);
}

extern "C" void kernel_launch(void* const* d_in, const int* in_sizes, int n_in,
                              void* d_out, int out_size, void* d_ws, size_t ws_size,
                              hipStream_t stream) {
  (void)in_sizes; (void)n_in; (void)out_size; (void)ws_size;

  const float* x     = (const float*)d_in[0];
  const int*   ei    = (const int*)d_in[1];
  const int*   srcI  = ei;
  const int*   dstI  = ei + EE;
  const int*   label = (const int*)d_in[2];
  const float* Wq_s  = (const float*)d_in[3];
  const float* bq_s  = (const float*)d_in[4];
  const float* Wk_s  = (const float*)d_in[5];
  const float* bk_s  = (const float*)d_in[6];
  const float* Wv_s  = (const float*)d_in[7];
  const float* bv_s  = (const float*)d_in[8];
  const float* Ws_s  = (const float*)d_in[9];
  const float* bs_s  = (const float*)d_in[10];
  const float* ln_w  = (const float*)d_in[11];
  const float* ln_b  = (const float*)d_in[12];
  const float* Wq_o  = (const float*)d_in[13];
  const float* bq_o  = (const float*)d_in[14];
  const float* Wk_o  = (const float*)d_in[15];
  const float* bk_o  = (const float*)d_in[16];
  const float* Wv_o  = (const float*)d_in[17];
  const float* bv_o  = (const float*)d_in[18];
  const float* Ws_o  = (const float*)d_in[19];
  const float* bs_o  = (const float*)d_in[20];
  const float* Vp    = (const float*)d_in[21];
  const float* bVp   = (const float*)d_in[22];
  const float* Up    = (const float*)d_in[23];
  const float* bUp   = (const float*)d_in[24];
  const float* wp    = (const float*)d_in[25];

  float* out = (float*)d_out;

  // workspace carve-up (256B aligned)
  char*  ws  = (char*)d_ws;
  size_t off = 0;
  auto alloc = [&](size_t bytes) -> void* {
    void* p = ws + off; off += (bytes + 255) & ~(size_t)255; return p;
  };
  unsigned short* hb    = (unsigned short*)alloc((size_t)NN * LL * 2);
  unsigned short* wbuf  = (unsigned short*)alloc((size_t)LL * LL * 2);
  float* qbuf  = (float*)alloc((size_t)NN * LL * 4);
  float* kbuf  = (float*)alloc((size_t)NN * LL * 4);
  float* vbuf  = (float*)alloc((size_t)NN * LL * 4);
  float* hnew  = (float*)alloc((size_t)NN * LL * 4);
  float* hcur  = (float*)alloc((size_t)NN * LL * 4);
  float* logits= (float*)alloc((size_t)EE * HH * 4);
  float* ewb   = (float*)alloc((size_t)EE * HH * 4);
  float* mbuf  = (float*)alloc((size_t)NN * HH * 4);
  float* sbuf  = (float*)alloc((size_t)NN * HH * 4);
  float* scal  = (float*)alloc(64 * 4);
  float* ho    = (float*)alloc((size_t)NN * DD * 4);
  unsigned short* hob = (unsigned short*)alloc((size_t)NN * DD * 2);
  float* gv    = (float*)alloc((size_t)NN * DD * 4);
  float* gu    = (float*)alloc((size_t)NN * DD * 4);
  float* glog  = (float*)alloc((size_t)NN * NCLSC * 4);
  float* attn  = (float*)alloc((size_t)NN * NCLSC * 4);

  const int nl    = NN * LL;                 // 10,240,000
  const int gNL   = (nl + 255) / 256;
  const int gLL2  = (LL * LL + 255) / 256;   // weight cast blocks
  const int eh    = EE * HH;
  const int gEH   = (eh + 255) / 256;
  const int gWaveEH = (eh + 7) / 8;          // one wave per (edge,head)
  const int nh    = NN * HH;
  const int gNH   = (nh + 255) / 256;

  // h = x
  copy_f32_kernel<<<gNL, 256, 0, stream>>>(x, hcur, nl);

  // ----- 4 stacked TransformerConv layers (hc=1024, heads=4, c=256) -----
  for (int i = 0; i < 4; ++i) {
    cast_f32_to_bf16_kernel<<<gNL, 256, 0, stream>>>(hcur, hb, nl);

    cast_f32_to_bf16_kernel<<<gLL2, 256, 0, stream>>>(Wq_s + (size_t)i * LL * LL, wbuf, LL * LL);
    launch_gemm(hb, wbuf, bq_s + (size_t)i * LL, qbuf, NN, LL, LL, stream);
    cast_f32_to_bf16_kernel<<<gLL2, 256, 0, stream>>>(Wk_s + (size_t)i * LL * LL, wbuf, LL * LL);
    launch_gemm(hb, wbuf, bk_s + (size_t)i * LL, kbuf, NN, LL, LL, stream);
    cast_f32_to_bf16_kernel<<<gLL2, 256, 0, stream>>>(Wv_s + (size_t)i * LL * LL, wbuf, LL * LL);
    launch_gemm(hb, wbuf, bv_s + (size_t)i * LL, vbuf, NN, LL, LL, stream);
    cast_f32_to_bf16_kernel<<<gLL2, 256, 0, stream>>>(Ws_s + (size_t)i * LL * LL, wbuf, LL * LL);
    launch_gemm(hb, wbuf, bs_s + (size_t)i * LL, hnew, NN, LL, LL, stream);  // skip term

    edge_logits_kernel<<<gWaveEH, 256, 0, stream>>>(qbuf, kbuf, srcI, dstI, logits,
                                                    EE, HH, LL / HH, 0.0625f);
    init_seg_kernel<<<gNH, 256, 0, stream>>>(mbuf, sbuf, nh);
    seg_max_kernel<<<gEH, 256, 0, stream>>>(logits, dstI, mbuf, EE, HH);
    m_fix_kernel<<<gNH, 256, 0, stream>>>(mbuf, nh);
    edge_exp_kernel<<<gEH, 256, 0, stream>>>(logits, mbuf, dstI, ewb, sbuf, EE, HH);
    edge_aggregate_kernel<<<EE, 256, 0, stream>>>(ewb, sbuf, vbuf, srcI, dstI, hnew, HH, LL / HH);

    fill_f32_kernel<<<1, 64, 0, stream>>>(scal, 0.0f, 2);
    reduce_sum_kernel<<<1024, 256, 0, stream>>>(hnew, scal, nl);
    reduce_sq_kernel<<<1024, 256, 0, stream>>>(hnew, scal, nl);
    ln_apply_kernel<<<gNL, 256, 0, stream>>>(hnew, hcur, ln_w + (size_t)i * LL,
                                             ln_b + (size_t)i * LL, scal, nl, LL);
  }

  // ----- output TransformerConv (hc=256, heads=4, c=64) -----
  const int nd  = NN * DD;
  const int gND = (nd + 255) / 256;
  const int gW256 = (LL * HCO + 255) / 256;
  const int gW64k = (DD * DD + 255) / 256;

  cast_f32_to_bf16_kernel<<<gNL, 256, 0, stream>>>(hcur, hb, nl);
  cast_f32_to_bf16_kernel<<<gW256, 256, 0, stream>>>(Wq_o, wbuf, LL * HCO);
  launch_gemm(hb, wbuf, bq_o, qbuf, NN, HCO, LL, stream);
  cast_f32_to_bf16_kernel<<<gW256, 256, 0, stream>>>(Wk_o, wbuf, LL * HCO);
  launch_gemm(hb, wbuf, bk_o, kbuf, NN, HCO, LL, stream);
  cast_f32_to_bf16_kernel<<<gW256, 256, 0, stream>>>(Wv_o, wbuf, LL * HCO);
  launch_gemm(hb, wbuf, bv_o, vbuf, NN, HCO, LL, stream);
  cast_f32_to_bf16_kernel<<<gW256, 256, 0, stream>>>(Ws_o, wbuf, LL * HCO);
  launch_gemm(hb, wbuf, bs_o, ho, NN, HCO, LL, stream);

  edge_logits_kernel<<<gWaveEH, 256, 0, stream>>>(qbuf, kbuf, srcI, dstI, logits,
                                                  EE, HH, HCO / HH, 0.125f);
  init_seg_kernel<<<gNH, 256, 0, stream>>>(mbuf, sbuf, nh);
  seg_max_kernel<<<gEH, 256, 0, stream>>>(logits, dstI, mbuf, EE, HH);
  m_fix_kernel<<<gNH, 256, 0, stream>>>(mbuf, nh);
  edge_exp_kernel<<<gEH, 256, 0, stream>>>(logits, mbuf, dstI, ewb, sbuf, EE, HH);
  edge_aggregate_kernel<<<EE, 256, 0, stream>>>(ewb, sbuf, vbuf, srcI, dstI, ho, HH, HCO / HH);

  gelu2_kernel<<<gND, 256, 0, stream>>>(ho, nd);

  // ----- gated pooling -----
  cast_f32_to_bf16_kernel<<<gND, 256, 0, stream>>>(ho, hob, nd);
  cast_f32_to_bf16_kernel<<<gW64k, 256, 0, stream>>>(Vp, wbuf, DD * DD);
  launch_gemm(hob, wbuf, bVp, gv, NN, DD, DD, stream);
  cast_f32_to_bf16_kernel<<<gW64k, 256, 0, stream>>>(Up, wbuf, DD * DD);
  launch_gemm(hob, wbuf, bUp, gu, NN, DD, DD, stream);
  gate_kernel<<<gND, 256, 0, stream>>>(gv, gu, nd);        // gv := g

  glog_kernel<<<(NN + 255) / 256, 256, 0, stream>>>(gv, wp, glog, NN, DD);
  softmax_init_kernel<<<1, 1, 0, stream>>>(scal);
  const int n2 = NN * NCLSC;
  colmax_kernel<<<(n2 + 255) / 256, 256, 0, stream>>>(glog, scal, n2);
  colexp_kernel<<<(n2 + 255) / 256, 256, 0, stream>>>(glog, attn, scal, n2);
  coldiv_kernel<<<(n2 + 255) / 256, 256, 0, stream>>>(attn, scal, n2);

  // outputs: y (2x256) | A (10000) | h (10000x256)
  fill_f32_kernel<<<2, 256, 0, stream>>>(out, 0.0f, NCLSC * DD);
  {
    dim3 g(40, NCLSC);
    yout_kernel<<<g, DD, 0, stream>>>(attn, ho, out, NN / 40);
  }
  aout_kernel<<<(NN + 255) / 256, 256, 0, stream>>>(attn, label, out + NCLSC * DD, NN);
  copy_f32_kernel<<<gND, 256, 0, stream>>>(ho, out + NCLSC * DD + NN, nd);
}
